// MS3_25829933318799
// MI455X (gfx1250) — matrix operand order repaired and checked by reading, not compiled, not run on hardware
//
#include <hip/hip_runtime.h>

// ---------------------------------------------------------------------------
// Hamiltonian/reversible net forward on MI455X (gfx1250).
//   per layer: Z -= H * tanh(Y@K1 + b2) @ K1^T ;  Y += H * tanh(Z@K2 + b1) @ K2^T
// v4: f32 Y/Z master state in registers (C-layout, wave-owned region); LDS has
// f16 shadows of Y/Z, tanh buffer, f16 weights (row- + col-major), and a raw
// f32 staging buffer fed by ASYNC global->LDS loads issued one phase ahead so
// the 64KB weight transfer overlaps the previous two GEMMs. WMMA inner loops
// are kc-outer/nt-inner with 4 independent accumulators and double-buffered
// operand fragments (no per-WMMA dscnt-0 stalls).
// ---------------------------------------------------------------------------

typedef _Float16 h16 __attribute__((ext_vector_type(16)));
typedef _Float16 h8  __attribute__((ext_vector_type(8)));
typedef _Float16 h4  __attribute__((ext_vector_type(4)));
typedef float    f8  __attribute__((ext_vector_type(8)));

#define NLAYERS 64
#define NHH     128
#define HSTEP   (1.0f / 64.0f)
#define HSR     136      // f16 LDS row stride (halves) -> conflict-free tile gathers
#define NT      512      // threads per block (16 waves, wave32)
#define MROWS   128      // batch rows per block

__device__ __forceinline__ float fast_tanh(float x) {
#if __has_builtin(__builtin_amdgcn_tanhf)
    return __builtin_amdgcn_tanhf(x);      // v_tanh_f32
#else
    float e = __expf(2.0f * x);
    return 1.0f - 2.0f / (e + 1.0f);
#endif
}

// Issue async DMA of one 128x128 f32 weight (64KB) global -> LDS staging.
// Tracked by ASYNCcnt; no VGPR data held in flight.
__device__ __forceinline__ void issue_async_weight(const float* __restrict__ Kg,
                                                   float* stage, int tid) {
#pragma unroll
    for (int t = 0; t < 8; ++t) {
        int idx4 = tid + t * NT;                              // 0..4095 float4s
        unsigned int       lds = (unsigned int)(size_t)(stage + (size_t)idx4 * 4);
        unsigned long long ga  = (unsigned long long)(Kg + (size_t)idx4 * 4);
        asm volatile("global_load_async_to_lds_b128 %0, %1, off"
                     :: "v"(lds), "v"(ga) : "memory");
    }
}

__device__ __forceinline__ void wait_async_done() {
    asm volatile("s_wait_asynccnt 0" ::: "memory");
}

// Convert staged f32 weight -> f16 row-major (rm) + col-major (cm) LDS.
__device__ __forceinline__ void convert_weights(const float* stage,
                                                _Float16* rm, _Float16* cm, int tid) {
#pragma unroll
    for (int t = 0; t < 8; ++t) {
        int idx4 = tid + t * NT;
        int r    = idx4 >> 5;                  // 32 float4 per row
        int c    = (idx4 & 31) << 2;
        float4 v = *(const float4*)(stage + (size_t)idx4 * 4);
        h4 p;
        p[0] = (_Float16)v.x; p[1] = (_Float16)v.y;
        p[2] = (_Float16)v.z; p[3] = (_Float16)v.w;
        *(h4*)&rm[r * HSR + c] = p;
        cm[(c + 0) * HSR + r] = p[0];
        cm[(c + 1) * HSR + r] = p[1];
        cm[(c + 2) * HSR + r] = p[2];
        cm[(c + 3) * HSR + r] = p[3];
    }
}

// A fragment, 16x32 f16: lane half kho -> K k0..k0+7 in elems 0..7, k0+16..k0+23
// in elems 8..15, where k0 = kc*32 + kho*8. Source is row-major f16 LDS.
__device__ __forceinline__ h16 load_afrag(const _Float16* Ah, int arow, int kc, int kho) {
    int k0 = kc * 32 + kho * 8;
    h8 lo = *(const h8*)&Ah[arow * HSR + k0];
    h8 hi = *(const h8*)&Ah[arow * HSR + k0 + 16];
    return __builtin_shufflevector(lo, hi, 0,1,2,3,4,5,6,7,8,9,10,11,12,13,14,15);
}

// B fragment, 32x16 f16: lane n holds K = kho*8 .. kho*8+7 and kho*8+16 .. kho*8+23
// of chunk kc, column-contiguous in the stored-major LDS buffer.
__device__ __forceinline__ h16 load_bfrag(const _Float16* Kb, int n, int kc, int kho) {
    int kb = kc * 32 + kho * 8;
    h8 lo = *(const h8*)&Kb[n * HSR + kb];
    h8 hi = *(const h8*)&Kb[n * HSR + kb + 16];
    return __builtin_shufflevector(lo, hi, 0,1,2,3,4,5,6,7,8,9,10,11,12,13,14,15);
}

// 16x64 output region GEMM over K=128 with software-pipelined operand feed.
__device__ __forceinline__ void gemm_128(const _Float16* Ah, const _Float16* Kb,
                                         int arow, int n0, int kho, f8 acc[4]) {
    h16 acur = load_afrag(Ah, arow, 0, kho);
    h16 bcur[4];
#pragma unroll
    for (int nti = 0; nti < 4; ++nti) bcur[nti] = load_bfrag(Kb, n0 + nti * 16, 0, kho);

#pragma unroll
    for (int kc = 0; kc < 4; ++kc) {
        h16 anext;
        h16 bnext[4];
        if (kc < 3) {                        // next round's loads go out first
            anext = load_afrag(Ah, arow, kc + 1, kho);
#pragma unroll
            for (int nti = 0; nti < 4; ++nti)
                bnext[nti] = load_bfrag(Kb, n0 + nti * 16, kc + 1, kho);
        }
#pragma unroll
        for (int nti = 0; nti < 4; ++nti)    // 4 independent WMMA chains
            acc[nti] = __builtin_amdgcn_wmma_f32_16x16x32_f16(
                false, acur, false, bcur[nti], (short)0, acc[nti], false, false);
        if (kc < 3) {
            acur = anext;
#pragma unroll
            for (int nti = 0; nti < 4; ++nti) bcur[nti] = bnext[nti];
        }
    }
}

__global__ __launch_bounds__(NT, 1)
void MS3_hamnet_kernel(const float* __restrict__ Y0,
                       const float* __restrict__ K1,
                       const float* __restrict__ K2,
                       const float* __restrict__ b1,
                       const float* __restrict__ b2,
                       float* __restrict__ out) {
    __shared__ _Float16 sYh[MROWS * HSR];      // 34.8 KB  f16 shadow of Y
    __shared__ _Float16 sZh[MROWS * HSR];      // 34.8 KB  f16 shadow of Z
    __shared__ _Float16 sT [MROWS * HSR];      // 34.8 KB  tanh output
    __shared__ _Float16 sKrm[NHH * HSR];       // 34.8 KB  K (f16, row-major)
    __shared__ _Float16 sKcm[NHH * HSR];       // 34.8 KB  K (f16, col-major)
    __shared__ float    sStage[NHH * NHH];     // 64.0 KB  raw f32 async staging
    __shared__ float    sB1[NHH];
    __shared__ float    sB2[NHH];

    const int tid  = threadIdx.x;
    const int wave = tid >> 5;
    const int lane = tid & 31;
    const int blk  = blockIdx.x;

    const int m0   = (wave >> 1) * 16;         // wave's 16-row band
    const int nt0  = (wave & 1) * 4;           // wave's 4 column tiles
    const int ln   = lane & 15;
    const int kho  = lane >> 4;
    const int arow = m0 + ln;
    const int nb   = nt0 * 16 + ln;            // wave's base output column per lane

    const f8 vzero = {0.f, 0.f, 0.f, 0.f, 0.f, 0.f, 0.f, 0.f};

    // f32 master state in registers, C-layout:
    //   yreg[nti][r] == Y[m0 + r + 8*kho][(nt0+nti)*16 + ln]
    float yreg[4][8];
    float zreg[4][8];

    // kick off DMA of layer 0's K1 while we set up state
    issue_async_weight(K1, sStage, tid);

    // ---------------- load initial Y | Z into registers + f16 shadows ----------------
    {
        const float* src = Y0 + (size_t)blk * MROWS * 256;
#pragma unroll
        for (int nti = 0; nti < 4; ++nti) {
            int col = nb + nti * 16;
#pragma unroll
            for (int r = 0; r < 8; ++r) {
                int row = m0 + r + 8 * kho;
                float yv = src[(size_t)row * 256 + col];
                float zv = src[(size_t)row * 256 + NHH + col];
                yreg[nti][r] = yv;
                zreg[nti][r] = zv;
                sYh[row * HSR + col] = (_Float16)yv;
                sZh[row * HSR + col] = (_Float16)zv;
            }
        }
    }

    for (int j = 0; j < NLAYERS; ++j) {
        const float* K2j = K2 + (size_t)j * NHH * NHH;
        const float* K1n = (j + 1 < NLAYERS) ? K1 + (size_t)(j + 1) * NHH * NHH : nullptr;

        // ---- K1 staged -> f16 LDS; biases ----
        __syncthreads();                        // (A) shadows/prev layer done
        wait_async_done();                      //     K1j DMA complete (per wave)
        convert_weights(sStage, sKrm, sKcm, tid);
        if (tid < NHH) {
            sB1[tid] = b1[(size_t)j * NHH + tid];
            sB2[tid] = b2[(size_t)j * NHH + tid];
        }
        __syncthreads();                        // (B) weights ready, staging free

        issue_async_weight(K2j, sStage, tid);   // K2 DMA overlaps GEMM1+2

        // ---- GEMM 1: sT = tanh(Yh @ K1 + b2) ----
        {
            float bv[4];
#pragma unroll
            for (int nti = 0; nti < 4; ++nti) bv[nti] = sB2[nb + nti * 16];
            f8 acc[4];
#pragma unroll
            for (int i = 0; i < 4; ++i) acc[i] = vzero;
            gemm_128(sYh, sKcm, arow, nb, kho, acc);
#pragma unroll
            for (int nti = 0; nti < 4; ++nti) {
                int n = nb + nti * 16;
#pragma unroll
                for (int r = 0; r < 8; ++r) {
                    float tv = fast_tanh(acc[nti][r] + bv[nti]);
                    sT[(m0 + r + 8 * kho) * HSR + n] = (_Float16)tv;
                }
            }
        }
        __syncthreads();                        // (C)

        // ---- GEMM 2: Z -= H * sT @ K1^T -> regs + sZh ----
        {
            f8 acc[4];
#pragma unroll
            for (int i = 0; i < 4; ++i) acc[i] = vzero;
            gemm_128(sT, sKrm, arow, nb, kho, acc);
#pragma unroll
            for (int nti = 0; nti < 4; ++nti) {
                int n = nb + nti * 16;
#pragma unroll
                for (int r = 0; r < 8; ++r) {
                    float zv = fmaf(-HSTEP, acc[nti][r], zreg[nti][r]);
                    zreg[nti][r] = zv;
                    sZh[(m0 + r + 8 * kho) * HSR + n] = (_Float16)zv;
                }
            }
        }
        __syncthreads();                        // (D) GEMM2 done, Krm/Kcm free

        // ---- K2 staged -> f16 LDS ----
        wait_async_done();
        convert_weights(sStage, sKrm, sKcm, tid);
        __syncthreads();                        // (E) weights ready, staging free

        if (K1n) issue_async_weight(K1n, sStage, tid);  // next layer's K1 DMA

        // ---- GEMM 3: sT = tanh(Zh @ K2 + b1) ----
        {
            float bv[4];
#pragma unroll
            for (int nti = 0; nti < 4; ++nti) bv[nti] = sB1[nb + nti * 16];
            f8 acc[4];
#pragma unroll
            for (int i = 0; i < 4; ++i) acc[i] = vzero;
            gemm_128(sZh, sKcm, arow, nb, kho, acc);
#pragma unroll
            for (int nti = 0; nti < 4; ++nti) {
                int n = nb + nti * 16;
#pragma unroll
                for (int r = 0; r < 8; ++r) {
                    float tv = fast_tanh(acc[nti][r] + bv[nti]);
                    sT[(m0 + r + 8 * kho) * HSR + n] = (_Float16)tv;
                }
            }
        }
        __syncthreads();                        // (F)

        // ---- GEMM 4: Y += H * sT @ K2^T -> regs + sYh ----
        {
            f8 acc[4];
#pragma unroll
            for (int i = 0; i < 4; ++i) acc[i] = vzero;
            gemm_128(sT, sKrm, arow, nb, kho, acc);
#pragma unroll
            for (int nti = 0; nti < 4; ++nti) {
                int n = nb + nti * 16;
#pragma unroll
                for (int r = 0; r < 8; ++r) {
                    float yv = fmaf(HSTEP, acc[nti][r], yreg[nti][r]);
                    yreg[nti][r] = yv;
                    sYh[(m0 + r + 8 * kho) * HSR + n] = (_Float16)yv;
                }
            }
        }
    }

    // ---------------- write out [Y | Z] straight from registers ----------------
    {
        float* dst = out + (size_t)blk * MROWS * 256;
#pragma unroll
        for (int nti = 0; nti < 4; ++nti) {
            int col = nb + nti * 16;
#pragma unroll
            for (int r = 0; r < 8; ++r) {
                int row = m0 + r + 8 * kho;
                dst[(size_t)row * 256 + col]       = yreg[nti][r];
                dst[(size_t)row * 256 + NHH + col] = zreg[nti][r];
            }
        }
    }
}

extern "C" void kernel_launch(void* const* d_in, const int* in_sizes, int n_in,
                              void* d_out, int out_size, void* d_ws, size_t ws_size,
                              hipStream_t stream) {
    const float* Y0 = (const float*)d_in[0];
    const float* K1 = (const float*)d_in[1];
    const float* K2 = (const float*)d_in[2];
    const float* b1 = (const float*)d_in[3];
    const float* b2 = (const float*)d_in[4];
    float* out      = (float*)d_out;

    const int batch = in_sizes[0] / 256;    // 32768
    const int nblk  = batch / MROWS;        // 256 workgroups

    hipLaunchKernelGGL(MS3_hamnet_kernel, dim3(nblk), dim3(NT), 0, stream,
                       Y0, K1, K2, b1, b2, out);
}
